// PoolingAttention_54116587929977
// MI455X (gfx1250) — matrix operand order repaired and checked by reading, not compiled, not run on hardware
//
#include <hip/hip_runtime.h>

typedef _Float16 half_t;
typedef __attribute__((ext_vector_type(16))) _Float16 v16h;
typedef __attribute__((ext_vector_type(8)))  _Float16 v8h;
typedef __attribute__((ext_vector_type(4)))  _Float16 v4h;
typedef __attribute__((ext_vector_type(4)))  float    v4f;
typedef __attribute__((ext_vector_type(8)))  float    v8f;

#define DIM      768
#define NHEADS   12
#define HDIM     64
#define SEQ      197
#define BATCH    64
#define KEEP     160
#define ATT_SCALE 0.125f           /* 64^-0.5 */
#define MROWS    (BATCH * SEQ)     /* 12608 */
#define QKV_COLS (3 * DIM)         /* 2304  */
#define KPAD     208               /* key rows padded (QK^T masking)   */
#define VPAD     224               /* key cols padded to 7*32 for PV   */

/* attention LDS layout (half elements, dynamic LDS only -> base offset 0) */
#define KSTR 72     /* K row stride: 144B, 16B-aligned, conflict-free */
#define VSTR 232    /* V^T row stride: 464B, 16B-aligned, conflict-free */
#define PSTR 40     /* P row stride: 80B */
#define LDS_K_HALFS (KPAD * KSTR)          /* 14976 */
#define LDS_V_HALFS (HDIM * VSTR)          /* 14848 */
#define LDS_P_HALFS (8 * 16 * PSTR)        /* 5120  */

/* GEMM LDS: block tile 64(M) x 256(N), double buffered */
#define GA_STR 40
#define GEMM_A_HALFS (64 * GA_STR)                  /* 2560  */
#define GEMM_B_HALFS (256 * GA_STR)                 /* 10240 */
#define GEMM_BUF_HALFS (GEMM_A_HALFS + GEMM_B_HALFS)/* 12800 -> 2 bufs = 51200B */

/* 16B global -> LDS async copy; ldsOff is allocation-relative byte offset. */
static __device__ __forceinline__ void async_ld16(unsigned int ldsOff, const half_t* g) {
  asm volatile("global_load_async_to_lds_b128 %0, %1, off"
               :: "v"(ldsOff), "v"(g) : "memory");
}

static __device__ __forceinline__ v16h ld_frag16(const half_t* p0, const half_t* p1) {
  v8h a = *(const v8h*)p0;
  v8h b = *(const v8h*)p1;
  v16h r;
#pragma unroll
  for (int i = 0; i < 8; ++i) { r[i] = a[i]; r[8 + i] = b[i]; }
  return r;
}

/* vectorized f32 -> f16 conversion (n4 = count/4) */
__global__ __launch_bounds__(256) void cvt_f16_kernel(const float* __restrict__ s,
                                                      half_t* __restrict__ d, int n4) {
  int i = blockIdx.x * 256 + threadIdx.x;
  if (i < n4) {
    v4f x = *(const v4f*)(s + 4 * (size_t)i);
    v4h y;
#pragma unroll
    for (int j = 0; j < 4; ++j) y[j] = (half_t)x[j];
    *(v4h*)(d + 4 * (size_t)i) = y;
  }
}

/* stage one 64x32 A tile + 256x32 W tile into LDS buffer (5 asyncs / thread) */
static __device__ __forceinline__ void gemm_stage(
    const half_t* __restrict__ A, const half_t* __restrict__ W, int K,
    int R0, int C0, int k0, int tid, unsigned int bufBase /* halfs */) {
  int r = tid >> 2, c = tid & 3;
  async_ld16((unsigned int)((bufBase + r * GA_STR + c * 8) * sizeof(half_t)),
             A + (size_t)(R0 + r) * K + k0 + c * 8);
#pragma unroll
  for (int i = 0; i < 4; ++i) {
    int idx = tid + i * 256, n = idx >> 2, cc = idx & 3;
    async_ld16((unsigned int)((bufBase + GEMM_A_HALFS + n * GA_STR + cc * 8) * sizeof(half_t)),
               W + (size_t)(C0 + n) * K + k0 + cc * 8);
  }
}

/* C = A[M x K] @ W[Nd x K]^T, f16 in, f32 accum.
   Block tile 64x256; 8 waves (wm 0..1 = 32 rows, wn 0..3 = 64 cols);
   each wave 2x4 16x16 tiles: 8 wmma per 12 ds_load_b128.
   mode 0: Cout(f32) = acc + bias.
   mode 1: qkv split epilogue -> f16 Qf[b,h,n,d] / Kf[b,h,208,d] / Vt[b,h,d,224]. */
__global__ __launch_bounds__(256) void wmma_gemm_f16(
    const half_t* __restrict__ A, const half_t* __restrict__ W, int Nd, int K,
    int mode, float* __restrict__ Cout, const float* __restrict__ bias,
    half_t* __restrict__ Qf, half_t* __restrict__ Kf, half_t* __restrict__ Vt) {
  extern __shared__ __align__(16) half_t smem[];

  const int tid = threadIdx.x;
  const int wave = tid >> 5, lane = tid & 31;
  const int ln = lane & 15, hi = lane >> 4, kb = hi ? 8 : 0;
  const int wm = wave & 1, wn = wave >> 1;
  const int R0 = blockIdx.x * 64;
  const int C0 = blockIdx.y * 256;

  v8f acc[2][4] = {};
  const int steps = K >> 5;

  gemm_stage(A, W, K, R0, C0, 0, tid, 0u);
  for (int ks = 0; ks < steps; ++ks) {
    unsigned int cur = (ks & 1) ? GEMM_BUF_HALFS : 0u;
    if (ks + 1 < steps) {
      gemm_stage(A, W, K, R0, C0, (ks + 1) << 5, tid,
                 (ks & 1) ? 0u : (unsigned int)GEMM_BUF_HALFS);
      asm volatile("s_wait_asynccnt 0x5" ::: "memory");  /* drain previous stage */
    } else {
      asm volatile("s_wait_asynccnt 0x0" ::: "memory");
    }
    __syncthreads();
    const half_t* Asb = smem + cur;
    const half_t* Bsb = smem + cur + GEMM_A_HALFS;
    v16h af[2];
#pragma unroll
    for (int mi = 0; mi < 2; ++mi) {
      const half_t* ap = Asb + (wm * 32 + mi * 16 + ln) * GA_STR;
      af[mi] = ld_frag16(ap + kb, ap + 16 + kb);
    }
    v16h bf[4];
#pragma unroll
    for (int t = 0; t < 4; ++t) {
      const half_t* bp = Bsb + (wn * 64 + t * 16 + ln) * GA_STR;
      bf[t] = ld_frag16(bp + kb, bp + 16 + kb);
    }
#pragma unroll
    for (int mi = 0; mi < 2; ++mi)
#pragma unroll
      for (int t = 0; t < 4; ++t)
        acc[mi][t] = __builtin_amdgcn_wmma_f32_16x16x32_f16(
            false, af[mi], false, bf[t], (short)0, acc[mi][t], false, false);
    __syncthreads();
  }

  if (mode == 0) {
#pragma unroll
    for (int t = 0; t < 4; ++t) {
      int col = C0 + wn * 64 + t * 16 + ln;
      float bv = bias[col];
#pragma unroll
      for (int mi = 0; mi < 2; ++mi)
#pragma unroll
        for (int j = 0; j < 8; ++j) {
          int row = R0 + wm * 32 + mi * 16 + j + 8 * hi;
          Cout[(size_t)row * Nd + col] = acc[mi][t][j] + bv;
        }
    }
  } else {
    const int s = C0 / DIM;              /* uniform: 256 | 768 */
    const int b0 = R0 / SEQ;             /* one uniform division */
    const int r0 = R0 - b0 * SEQ;
#pragma unroll
    for (int t = 0; t < 4; ++t) {
      int col = C0 + wn * 64 + t * 16 + ln;
      int rem = col - s * DIM, hh = rem >> 6, d = rem & 63;
#pragma unroll
      for (int mi = 0; mi < 2; ++mi)
#pragma unroll
        for (int j = 0; j < 8; ++j) {
          int n = r0 + wm * 32 + mi * 16 + j + 8 * hi; /* < 2*SEQ: one wrap max */
          int bb = b0;
          if (n >= SEQ) { n -= SEQ; bb += 1; }
          int bh = bb * NHEADS + hh;
          half_t v = (half_t)acc[mi][t][j];
          if (s == 0)      Qf[((size_t)bh * SEQ + n) * HDIM + d] = v;
          else if (s == 1) Kf[((size_t)bh * KPAD + n) * HDIM + d] = v;
          else             Vt[((size_t)bh * HDIM + d) * VPAD + n] = v;
        }
    }
  }
}

/* One block per (b,h). Full attention over 197 tokens, head dim 64. */
__global__ __launch_bounds__(256) void attn_kernel(
    const half_t* __restrict__ Qf, const half_t* __restrict__ Kf,
    const half_t* __restrict__ Vt, half_t* __restrict__ attnout,
    float* __restrict__ row0w) {
  extern __shared__ __align__(16) half_t smem[];
  half_t* Kl = smem;                    /* [208][KSTR] */
  half_t* Pl = smem + LDS_K_HALFS + LDS_V_HALFS;

  const int bh = blockIdx.x;
  const int b = bh / NHEADS, h = bh % NHEADS;
  const int tid = threadIdx.x;
  const int wave = tid >> 5, lane = tid & 31;
  const int ln = lane & 15, hi = lane >> 4, kb = hi ? 8 : 0;

  /* async-stage K [208][64] and V^T [64][224] (both zero-padded in global) */
  const int KCH = KPAD * 8;             /* 1664 16B chunks */
  const int VCH = HDIM * 28;            /* 1792 16B chunks */
  for (int idx = tid; idx < KCH + VCH; idx += 256) {
    if (idx < KCH) {
      int r = idx >> 3, c = idx & 7;
      async_ld16((unsigned int)((r * KSTR + c * 8) * sizeof(half_t)),
                 Kf + ((size_t)bh * KPAD + r) * HDIM + c * 8);
    } else {
      int j = idx - KCH; int d = j / 28, c = j - d * 28;
      async_ld16((unsigned int)((LDS_K_HALFS + d * VSTR + c * 8) * sizeof(half_t)),
                 Vt + ((size_t)bh * HDIM + d) * VPAD + c * 8);
    }
  }
  asm volatile("s_wait_asynccnt 0x0" ::: "memory");
  __syncthreads();

  half_t* pw = Pl + wave * (16 * PSTR);
  const half_t* Vl = smem + LDS_K_HALFS;

  for (int rt = wave; rt < 13; rt += 8) {
    int qrow = rt * 16 + ln; if (qrow > SEQ - 1) qrow = SEQ - 1;
    const half_t* qp = Qf + ((size_t)bh * SEQ + qrow) * HDIM;
    v16h qa[2];
#pragma unroll
    for (int f = 0; f < 2; ++f)
      qa[f] = ld_frag16(qp + f * 32 + kb, qp + f * 32 + 16 + kb);

    /* S = Q K^T */
    v8f S[13];
#pragma unroll
    for (int t = 0; t < 13; ++t) {
      v8f c = {};
#pragma unroll
      for (int f = 0; f < 2; ++f) {
        const half_t* kp = Kl + (t * 16 + ln) * KSTR + f * 32;
        v16h bf = ld_frag16(kp + kb, kp + 16 + kb);
        c = __builtin_amdgcn_wmma_f32_16x16x32_f16(false, qa[f], false, bf,
                                                   (short)0, c, false, false);
      }
      S[t] = c;
    }
#pragma unroll
    for (int t = 0; t < 13; ++t) {
      int key = t * 16 + ln;
#pragma unroll
      for (int j = 0; j < 8; ++j) {
        float sv = S[t][j] * ATT_SCALE;
        S[t][j] = (key < SEQ) ? sv : -1e30f;
      }
    }
    if (rt == 0 && hi == 0) {
#pragma unroll
      for (int t = 0; t < 13; ++t) {
        int key = t * 16 + ln;
        if (key < SEQ) row0w[((size_t)b * NHEADS + h) * SEQ + key] = fabsf(S[t][0]);
      }
    }
    /* softmax across the 16 lanes holding each row */
    float lrow[8];
#pragma unroll
    for (int j = 0; j < 8; ++j) {
      float mx = -1e30f;
#pragma unroll
      for (int t = 0; t < 13; ++t) mx = fmaxf(mx, S[t][j]);
#pragma unroll
      for (int m = 8; m >= 1; m >>= 1) mx = fmaxf(mx, __shfl_xor(mx, m, 32));
      float sum = 0.0f;
#pragma unroll
      for (int t = 0; t < 13; ++t) { float p = __expf(S[t][j] - mx); S[t][j] = p; sum += p; }
#pragma unroll
      for (int m = 8; m >= 1; m >>= 1) sum += __shfl_xor(sum, m, 32);
      lrow[j] = sum;
    }
    /* O = P V (keys padded to 224; tile 13 written as zeros) */
    v8f O[4] = {};
#pragma unroll
    for (int s = 0; s < 7; ++s) {
#pragma unroll
      for (int j = 0; j < 8; ++j) {
        int r = j + 8 * hi;
        pw[r * PSTR + ln]      = (half_t)S[2 * s][j];
        pw[r * PSTR + 16 + ln] = (half_t)((2 * s + 1 < 13) ? S[2 * s + 1][j] : 0.0f);
      }
      asm volatile("s_wait_dscnt 0x0" ::: "memory");
      const half_t* pp = pw + ln * PSTR;
      v16h pa = ld_frag16(pp + kb, pp + 16 + kb);
#pragma unroll
      for (int dt = 0; dt < 4; ++dt) {
        const half_t* vp = Vl + (dt * 16 + ln) * VSTR + 32 * s;
        v16h bf = ld_frag16(vp + kb, vp + 16 + kb);
        O[dt] = __builtin_amdgcn_wmma_f32_16x16x32_f16(false, pa, false, bf,
                                                       (short)0, O[dt], false, false);
      }
    }
#pragma unroll
    for (int dt = 0; dt < 4; ++dt) {
#pragma unroll
      for (int j = 0; j < 8; ++j) {
        int row = rt * 16 + j + 8 * hi;
        if (row < SEQ)
          attnout[((size_t)b * SEQ + row) * DIM + h * HDIM + dt * 16 + ln] =
              (half_t)(O[dt][j] / lrow[j]);
      }
    }
  }
}

/* Per batch: sum |CLS scores| over heads, keep top-160 (jax tie-break),
   emit kept indices ascending. */
__global__ __launch_bounds__(256) void topk_kernel(const float* __restrict__ row0w,
                                                   int* __restrict__ keep_out) {
  __shared__ float w[SEQ];
  __shared__ unsigned char keep[SEQ];
  const int b = blockIdx.x;
  const int t = threadIdx.x;
  if (t < SEQ) {
    float s = 0.0f;
    for (int h = 0; h < NHEADS; ++h) s += row0w[((size_t)b * NHEADS + h) * SEQ + t];
    w[t] = s;
  }
  __syncthreads();
  if (t < SEQ) {
    float wi = w[t];
    int rank = 0;
    for (int j = 0; j < SEQ; ++j) {
      float wj = w[j];
      rank += (wj > wi) || (wj == wi && j < t);
    }
    keep[t] = (rank < KEEP) ? 1 : 0;
  }
  __syncthreads();
  if (t == 0) {
    int pos = 0;
    for (int j = 0; j < SEQ; ++j)
      if (keep[j]) keep_out[(size_t)b * KEEP + (pos++)] = j;
  }
}

extern "C" void kernel_launch(void* const* d_in, const int* in_sizes, int n_in,
                              void* d_out, int out_size, void* d_ws, size_t ws_size,
                              hipStream_t stream) {
  (void)in_sizes; (void)n_in; (void)out_size; (void)ws_size;
  const float* x      = (const float*)d_in[0];
  const float* qkv_w  = (const float*)d_in[1];
  const float* proj_w = (const float*)d_in[2];
  const float* proj_b = (const float*)d_in[3];

  float* out      = (float*)d_out;
  int*   keep_out = (int*)(out + (size_t)MROWS * DIM);

  /* workspace (halfs unless noted) */
  half_t* x16   = (half_t*)d_ws;                                 /* 12608*768   */
  half_t* qw16  = x16  + (size_t)MROWS * DIM;                    /* 2304*768    */
  half_t* pw16  = qw16 + (size_t)QKV_COLS * DIM;                 /* 768*768     */
  half_t* Qf    = pw16 + (size_t)DIM * DIM;                      /* 768*197*64  */
  half_t* Kf    = Qf   + (size_t)BATCH * NHEADS * SEQ * HDIM;    /* 768*208*64  */
  half_t* Vt    = Kf   + (size_t)BATCH * NHEADS * KPAD * HDIM;   /* 768*64*224  */
  half_t* ao16  = Vt   + (size_t)BATCH * NHEADS * HDIM * VPAD;   /* 12608*768   */
  float*  row0w = (float*)(ao16 + (size_t)MROWS * DIM);          /* 64*12*197 f32 */

  /* 0) f32 -> f16 copies of inputs */
  cvt_f16_kernel<<<(MROWS * DIM) / 4 / 256, 256, 0, stream>>>(x, x16, (MROWS * DIM) / 4);
  cvt_f16_kernel<<<(QKV_COLS * DIM) / 4 / 256, 256, 0, stream>>>(qkv_w, qw16, (QKV_COLS * DIM) / 4);
  cvt_f16_kernel<<<(DIM * DIM) / 4 / 256, 256, 0, stream>>>(proj_w, pw16, (DIM * DIM) / 4);

  /* zero pad regions of Kf / Vt */
  hipMemsetAsync(Kf, 0, (size_t)BATCH * NHEADS * KPAD * HDIM * sizeof(half_t), stream);
  hipMemsetAsync(Vt, 0, (size_t)BATCH * NHEADS * HDIM * VPAD * sizeof(half_t), stream);

  size_t gemmLds = (size_t)(2 * GEMM_BUF_HALFS) * sizeof(half_t);   /* 51200B */

  /* 1) qkv GEMM with split f16 epilogue (Q / K / V^T layouts) */
  wmma_gemm_f16<<<dim3(MROWS / 64, QKV_COLS / 256), 256, gemmLds, stream>>>(
      x16, qw16, QKV_COLS, DIM, 1, nullptr, nullptr, Qf, Kf, Vt);

  /* 2) fused attention per (b,h) */
  size_t attLds = (size_t)(LDS_K_HALFS + LDS_V_HALFS + LDS_P_HALFS) * sizeof(half_t);
  hipFuncSetAttribute(reinterpret_cast<const void*>(attn_kernel),
                      hipFuncAttributeMaxDynamicSharedMemorySize, (int)attLds);
  attn_kernel<<<BATCH * NHEADS, 256, attLds, stream>>>(Qf, Kf, Vt, ao16, row0w);

  /* 3) top-160 kept token indices */
  topk_kernel<<<BATCH, 256, 0, stream>>>(row0w, keep_out);

  /* 4) out = attnout @ proj_w^T + proj_b */
  wmma_gemm_f16<<<dim3(MROWS / 64, DIM / 256), 256, gemmLds, stream>>>(
      ao16, pw16, DIM, DIM, 0, out, proj_b, nullptr, nullptr, nullptr);
}
